// IOCClassifier_18030272708871
// MI455X (gfx1250) — matrix operand order repaired and checked by reading, not compiled, not run on hardware
//
#include <hip/hip_runtime.h>

// ---------------------------------------------------------------------------
// Types for CDNA5 WMMA (wave32, v_wmma_f32_16x16x32_bf16)
// ---------------------------------------------------------------------------
typedef __bf16 bf16;
typedef __attribute__((ext_vector_type(16))) __bf16 v16bf;
typedef __attribute__((ext_vector_type(8)))  __bf16 v8bf;
typedef __attribute__((ext_vector_type(8)))  float  v8f;
typedef __attribute__((ext_vector_type(4)))  float  v4f;

// Problem constants (from reference)
constexpr int N_   = 50000;
constexpr int E_   = 300000;
constexpr int DIN  = 128;
constexpr int H    = 256;
constexpr int R    = 4;
constexpr int Bb   = 16;
constexpr int Ll   = 4;
constexpr int ETE  = 16;
constexpr int KC   = R * H + H;   // 1280: [rel_mean | h] concat K-dim
constexpr int TILES = N_ / 16;    // 3125 row tiles of 16

// B staging chunk: 32 K-values (1 WMMA K-step) for all 256 output columns,
// double-buffered in LDS: 2 * 256 * (32+8) * 2B = 40 KB.
constexpr int CHUNK_K    = 32;
constexpr int LDS_STRIDE = CHUNK_K + 8;           // 80B/col -> conflict-free reads
constexpr int LDS_ELEMS  = H * LDS_STRIDE;        // one buffer

// ---------------------------------------------------------------------------
// Utility kernels
// ---------------------------------------------------------------------------
__global__ void k_zero(float* __restrict__ p, long n) {
    long i = (long)blockIdx.x * blockDim.x + threadIdx.x;
    long st = (long)gridDim.x * blockDim.x;
    for (; i < n; i += st) p[i] = 0.0f;
}

// wt[o][k] = (bf16) w[k][o]   (B operand wants contiguous K per output column)
__global__ void k_wt(const float* __restrict__ w, bf16* __restrict__ wt, int K, int Ncols) {
    int tid = blockIdx.x * blockDim.x + threadIdx.x;
    if (tid >= K * Ncols) return;
    int o = tid / K, k = tid - o * K;
    wt[(size_t)o * K + k] = (bf16)w[(size_t)k * Ncols + o];
}

// wcat[o][k] : k<1024 -> sum_b comp[r][b]*basis[b][i][o]  (r=k/256, i=k%256)
//              k>=1024 -> root_w[i][o]
__global__ void k_wcat(const float* __restrict__ basis_l, const float* __restrict__ comp_l,
                       const float* __restrict__ rootw_l, bf16* __restrict__ wcat) {
    int tid = blockIdx.x * blockDim.x + threadIdx.x;
    if (tid >= H * KC) return;
    int o = tid / KC, k = tid - o * KC;
    float v;
    if (k < R * H) {
        int r = k >> 8, i = k & 255;
        v = 0.0f;
        #pragma unroll
        for (int b = 0; b < Bb; ++b)
            v += comp_l[r * Bb + b] * basis_l[((size_t)b * H + i) * H + o];
    } else {
        int i = k - R * H;
        v = rootw_l[(size_t)i * H + o];
    }
    wcat[(size_t)o * KC + k] = (bf16)v;
}

// ---------------------------------------------------------------------------
// Edge MLP + degree / relation counts (one wave per edge, 8 cols per lane)
// ---------------------------------------------------------------------------
__global__ void k_edge(const int* __restrict__ ei, const int* __restrict__ etp,
                       const float* __restrict__ ew, const float* __restrict__ eemb,
                       const float* __restrict__ mw, const float* __restrict__ mb,
                       float* __restrict__ agg, float* __restrict__ deg, float* __restrict__ cnt) {
    int gt = blockIdx.x * blockDim.x + threadIdx.x;
    int e = gt >> 5, lane = gt & 31;
    if (e >= E_) return;
    int dst = ei[E_ + e];
    int rt  = etp[e];
    float w = ew[e];
    int c0 = lane * 8;
    float acc[8];
    {
        const v4f* bp = (const v4f*)(mb + c0);
        v4f b0 = bp[0], b1 = bp[1];
        #pragma unroll
        for (int j = 0; j < 4; ++j) { acc[j] = b0[j]; acc[4 + j] = b1[j]; }
    }
    #pragma unroll
    for (int k = 0; k < ETE; ++k) {
        float ek = eemb[rt * ETE + k];
        const v4f* wp = (const v4f*)(mw + (size_t)k * H + c0);
        v4f w0 = wp[0], w1 = wp[1];
        #pragma unroll
        for (int j = 0; j < 4; ++j) { acc[j] += ek * w0[j]; acc[4 + j] += ek * w1[j]; }
    }
    {   // last input row = edge weight
        const v4f* wp = (const v4f*)(mw + (size_t)ETE * H + c0);
        v4f w0 = wp[0], w1 = wp[1];
        #pragma unroll
        for (int j = 0; j < 4; ++j) { acc[j] += w * w0[j]; acc[4 + j] += w * w1[j]; }
    }
    float* ap = agg + (size_t)dst * H + c0;
    #pragma unroll
    for (int j = 0; j < 8; ++j) atomicAdd(ap + j, fmaxf(acc[j], 0.0f));
    if (lane == 0) {
        atomicAdd(&deg[dst], 1.0f);
        atomicAdd(&cnt[dst * R + rt], 1.0f);
    }
}

// rel_sum[(dst*R+rt)][:] += h[src][:]   (one wave per edge)
__global__ void k_scatter(const int* __restrict__ ei, const int* __restrict__ etp,
                          const float* __restrict__ h, float* __restrict__ rel) {
    int gt = blockIdx.x * blockDim.x + threadIdx.x;
    int e = gt >> 5, lane = gt & 31;
    if (e >= E_) return;
    int src = ei[e], dst = ei[E_ + e], rt = etp[e];
    int c0 = lane * 8;
    const v4f* hp = (const v4f*)(h + (size_t)src * H + c0);
    v4f a = hp[0], b = hp[1];
    float* rp = rel + ((size_t)dst * R + rt) * H + c0;
    #pragma unroll
    for (int j = 0; j < 4; ++j) { atomicAdd(rp + j, a[j]); atomicAdd(rp + 4 + j, b[j]); }
}

// ---------------------------------------------------------------------------
// Fused WMMA GEMM + (edge_term) + LayerNorm + ReLU + (residual)
//
// A: fp32 (rel_sum | h  or  x), scaled by 1/max(cnt,1) per relation block,
//    converted to bf16 in registers in the WMMA A layout.
// B: Wcat^T bf16, staged per 32-K chunk into *double-buffered* LDS with CDNA5
//    async global->LDS loads (ASYNCcnt). Chunk c+1 is issued before computing
//    chunk c; one barrier per chunk orders buffer reuse across waves.
//    Column tiles are fed through a distance-2 register pipeline (3 live B
//    buffers) so ds_loads are issued two WMMAs ahead of their consumer.
// C: f32 16x16 per wave; epilogue LN via __shfl_xor butterfly per half-wave.
// ---------------------------------------------------------------------------
__global__ __launch_bounds__(256, 1)
void k_gemm(const float* __restrict__ A0, int lda0, int K0,
            const float* __restrict__ cntp,
            const float* __restrict__ A1, int lda1, int Ktot,
            const bf16* __restrict__ Wt,
            const float* __restrict__ bias, const float* __restrict__ lng,
            const float* __restrict__ lnb,
            const float* __restrict__ agg, const float* __restrict__ deg,
            const float* __restrict__ hres,
            float* __restrict__ out, float* __restrict__ out2) {
    __shared__ __align__(16) bf16 ldsB[2 * LDS_ELEMS];   // 40 KB double buffer

    int lane = threadIdx.x & 31;
    int wave = threadIdx.x >> 5;
    int tile = blockIdx.x * 8 + wave;
    bool active = tile < TILES;           // uniform per wave -> EXEC stays all-ones
    int tt = active ? tile : 0;
    int l16 = lane & 15, g = lane >> 4;
    int row = tt * 16 + l16;              // A row owned by this lane

    const float* arow0 = A0 + (size_t)row * lda0;
    const float* arow1 = A1 ? (A1 + (size_t)row * lda1) : nullptr;

    float inv[4] = {1.0f, 1.0f, 1.0f, 1.0f};
    if (cntp) {
        #pragma unroll
        for (int r = 0; r < R; ++r) inv[r] = 1.0f / fmaxf(cntp[(size_t)row * R + r], 1.0f);
    }

    v8f acc[16];
    #pragma unroll
    for (int t = 0; t < 16; ++t)
        #pragma unroll
        for (int i = 0; i < 8; ++i) acc[t][i] = 0.0f;

    int nchunks = Ktot / CHUNK_K;
    int col = threadIdx.x;                                  // staging column 0..255

    // async-stage one 32-K chunk of B into buffer `sel`
    auto stage = [&](int c, int sel) {
        const bf16* gsrc = Wt + (size_t)col * Ktot + c * CHUNK_K;
        bf16* ldst = &ldsB[sel * LDS_ELEMS + col * LDS_STRIDE];
        #pragma unroll
        for (int i = 0; i < CHUNK_K / 8; ++i) {             // 4 x b128
            unsigned la = (unsigned)(uintptr_t)(ldst + i * 8);
            unsigned long long ga = (unsigned long long)(uintptr_t)(gsrc + i * 8);
            asm volatile("global_load_async_to_lds_b128 %0, %1, off"
                         :: "v"(la), "v"(ga) : "memory");
        }
    };

    stage(0, 0);                                            // prime the pipeline
    for (int c = 0; c < nchunks; ++c) {
        int sel = c & 1;
        // chunk c's transfers (issued last iteration) are the only outstanding ones
        asm volatile("s_wait_asynccnt 0x0" ::: "memory");
        __syncthreads();            // all data landed AND all waves done with buf sel^1
                                    // (their chunk c-1 compute preceded this barrier)
        if (c + 1 < nchunks) stage(c + 1, sel ^ 1);         // overlap with compute

        // ---- A fetch: fp32 -> scale -> bf16 in WMMA A layout ----
        int kglob = c * CHUNK_K;
        const float* ap; float sc;
        if (kglob < K0) { ap = arow0 + kglob;        sc = cntp ? inv[kglob >> 8] : 1.0f; }
        else            { ap = arow1 + (kglob - K0); sc = 1.0f; }
        __builtin_prefetch(ap + CHUNK_K, 0, 1);             // global_prefetch_b8
        const v4f* af0 = (const v4f*)(ap + 8 * g);
        v4f f0 = af0[0], f1 = af0[1];
        const v4f* af1 = (const v4f*)(ap + 16 + 8 * g);
        v4f f2 = af1[0], f3 = af1[1];
        v16bf a;
        #pragma unroll
        for (int i = 0; i < 4; ++i) {
            a[i]      = (bf16)(f0[i] * sc);
            a[4 + i]  = (bf16)(f1[i] * sc);
            a[8 + i]  = (bf16)(f2[i] * sc);
            a[12 + i] = (bf16)(f3[i] * sc);
        }

        // ---- 16 column tiles, distance-2 pipelined B feed from LDS ----
        const bf16* cur = &ldsB[sel * LDS_ELEMS];
        auto ldB = [&](int t) -> v16bf {
            const bf16* bp = cur + (l16 + 16 * t) * LDS_STRIDE + 8 * g;
            v8bf b0 = *(const v8bf*)bp;
            v8bf b1 = *(const v8bf*)(bp + 16);
            return __builtin_shufflevector(b0, b1,
                0,1,2,3,4,5,6,7,8,9,10,11,12,13,14,15);
        };
        v16bf bA = ldB(0);
        v16bf bB = ldB(1);
        #pragma unroll
        for (int t = 0; t < 16; ++t) {
            v16bf curb = (t & 1) ? bB : bA;
            if (t + 2 < 16) {                  // load 2 tiles ahead of the consumer
                v16bf bn = ldB(t + 2);
                if (t & 1) bB = bn; else bA = bn;
            }
            acc[t] = __builtin_amdgcn_wmma_f32_16x16x32_bf16(
                false, a, false, curb, (short)0, acc[t], false, false);
        }
    }

    // ---- epilogue ----
    if (!active) return;
    float biasv[16], gv[16], bvv[16];
    int colv[16];
    #pragma unroll
    for (int t = 0; t < 16; ++t) {
        int cc = l16 + 16 * t;
        colv[t] = cc; biasv[t] = bias[cc]; gv[t] = lng[cc]; bvv[t] = lnb[cc];
    }
    #pragma unroll
    for (int j = 0; j < 8; ++j) {
        int rowj = tt * 16 + j + 8 * g;
        float invd = 0.0f;
        if (agg) invd = 0.1f / fmaxf(deg[rowj], 1.0f);
        float w[16];
        #pragma unroll
        for (int t = 0; t < 16; ++t) {
            float v = acc[t][j] + biasv[t];
            if (agg) v += invd * agg[(size_t)rowj * H + colv[t]];
            w[t] = v;
        }
        float s = 0.0f, ss = 0.0f;
        #pragma unroll
        for (int t = 0; t < 16; ++t) { s += w[t]; ss += w[t] * w[t]; }
        #pragma unroll
        for (int m = 1; m < 16; m <<= 1) {          // butterfly over the 16-lane half
            s  += __shfl_xor(s,  m, 32);
            ss += __shfl_xor(ss, m, 32);
        }
        float mean = s * (1.0f / H);
        float var  = ss * (1.0f / H) - mean * mean;
        float rstd = rsqrtf(var + 1e-5f);
        #pragma unroll
        for (int t = 0; t < 16; ++t) {
            float y = (w[t] - mean) * rstd * gv[t] + bvv[t];
            y = fmaxf(y, 0.0f);
            if (hres) y += hres[(size_t)rowj * H + colv[t]];
            size_t idx = (size_t)rowj * H + colv[t];
            out[idx] = y;
            if (out2) out2[idx] = y;
        }
    }
}

// ---------------------------------------------------------------------------
// Host launcher
// ---------------------------------------------------------------------------
extern "C" void kernel_launch(void* const* d_in, const int* in_sizes, int n_in,
                              void* d_out, int out_size, void* d_ws, size_t ws_size,
                              hipStream_t stream) {
    const float* x      = (const float*)d_in[0];
    const int*   ei     = (const int*)  d_in[1];
    const int*   etp    = (const int*)  d_in[2];
    const float* ew     = (const float*)d_in[3];
    const float* proj_w = (const float*)d_in[4];
    const float* proj_b = (const float*)d_in[5];
    const float* pln_g  = (const float*)d_in[6];
    const float* pln_b  = (const float*)d_in[7];
    const float* eemb   = (const float*)d_in[8];
    const float* mlpw   = (const float*)d_in[9];
    const float* mlpb   = (const float*)d_in[10];
    const float* basis  = (const float*)d_in[11];
    const float* comp   = (const float*)d_in[12];
    const float* rootw  = (const float*)d_in[13];
    const float* rootb  = (const float*)d_in[14];
    const float* lng    = (const float*)d_in[15];
    const float* lnb    = (const float*)d_in[16];
    float* outp = (float*)d_out;
    (void)in_sizes; (void)n_in; (void)out_size; (void)ws_size;

    char* ws = (char*)d_ws;
    size_t off = 0;
    auto carve = [&](size_t bytes) -> char* {
        char* p = ws + off;
        off = (off + bytes + 255) & ~(size_t)255;
        return p;
    };
    float* hA   = (float*)carve((size_t)N_ * H * 4);
    float* hB   = (float*)carve((size_t)N_ * H * 4);   // double buffer (A read live)
    float* agg  = (float*)carve((size_t)N_ * H * 4);
    float* deg  = (float*)carve((size_t)N_ * 4);
    float* cnt  = (float*)carve((size_t)N_ * R * 4);
    float* rel  = (float*)carve((size_t)N_ * R * H * 4);
    bf16*  pwt  = (bf16*) carve((size_t)H * DIN * 2);
    bf16*  wcat = (bf16*) carve((size_t)H * KC * 2);

    const int TB = 256;
    // zero agg / deg / cnt (layer-invariant accumulators)
    k_zero<<<4096, TB, 0, stream>>>(agg, (long)N_ * H);
    k_zero<<<64,   TB, 0, stream>>>(deg, (long)N_);
    k_zero<<<64,   TB, 0, stream>>>(cnt, (long)N_ * R);

    // bf16 transpose of projection weight
    k_wt<<<(DIN * H + TB - 1) / TB, TB, 0, stream>>>(proj_w, pwt, DIN, H);

    // edge MLP -> agg/deg/cnt (atomic scatter)
    k_edge<<<E_ * 32 / TB, TB, 0, stream>>>(ei, etp, ew, eemb, mlpw, mlpb, agg, deg, cnt);

    // input projection: h = relu(LN(x @ proj_w + b))
    int gblocks = (TILES + 7) / 8;
    k_gemm<<<gblocks, TB, 0, stream>>>(x, DIN, DIN, nullptr, nullptr, 0, DIN, pwt,
                                       proj_b, pln_g, pln_b,
                                       nullptr, nullptr, nullptr, hA, nullptr);

    float* hin = hA;
    float* hout = hB;
    for (int l = 0; l < Ll; ++l) {
        k_zero<<<8192, TB, 0, stream>>>(rel, (long)N_ * R * H);
        k_scatter<<<E_ * 32 / TB, TB, 0, stream>>>(ei, etp, hin, rel);
        k_wcat<<<(H * KC + TB - 1) / TB, TB, 0, stream>>>(
            basis + (size_t)l * Bb * H * H, comp + (size_t)l * R * Bb,
            rootw + (size_t)l * H * H, wcat);
        k_gemm<<<gblocks, TB, 0, stream>>>(rel, R * H, R * H, cnt, hin, H, KC, wcat,
                                           rootb + (size_t)l * H,
                                           lng + (size_t)l * H, lnb + (size_t)l * H,
                                           agg, deg, /*hres=*/hin, /*out=*/hout,
                                           /*out2=*/(l == Ll - 1) ? outp : nullptr);
        float* tmp = hin; hin = hout; hout = tmp;
    }
}
